// FastThumbnail_89532888252587
// MI455X (gfx1250) — compile-verified, matrix operand order checked by reading, not run on hardware
//
#include <hip/hip_runtime.h>

typedef __attribute__((ext_vector_type(16))) _Float16 v16h;
typedef __attribute__((ext_vector_type(8)))  float    v8f;
typedef __attribute__((ext_vector_type(4)))  unsigned int u32x4;
typedef __attribute__((ext_vector_type(8)))  int      i32x8;
typedef __attribute__((ext_vector_type(4)))  int      i32x4;

#define NKS     6
#define NPAIR   18
#define NSRC    1024
#define SP      1126        // padded image side: 1024 + int(0.1*1024)
#define OUT_SLOT 40960      // >= max Hout*Wout = 35*1095
#define NCHUNK  128
#define CW      176         // NCHUNK + 48 halo (k <= 47)
#define NTILES  11          // CW / 16

struct PairTab {
    int k[NPAIR], HF[NPAIR], Hout[NPAIR], Wout[NPAIR];
    int covoff[NPAIR], repoff[NPAIR];
};

// ---------------------------------------------------------------------------
// Stage 1: build the three pooled feature images (f16 slabs, zero-padded).
// ---------------------------------------------------------------------------
__global__ void build_feats(const float* __restrict__ S, _Float16* __restrict__ featH)
{
    int li  = blockIdx.y;
    int pos = blockIdx.x * blockDim.x + threadIdx.x;
    if (pos >= SP * SP) return;
    int y = pos / SP, x = pos % SP;
    int HF = SP - li;                     // 1126, 1125, 1124
    float v = 0.f;
    if (y < HF && x < HF) {
        int p = li + 1;                   // pool size 1,2,3
        float s = 0.f;
        for (int dy = 0; dy <= li; ++dy)
            for (int dx = 0; dx <= li; ++dx) {
                int yy = y + dy, xx = x + dx;
                s += (yy < NSRC && xx < NSRC) ? S[yy * NSRC + xx] : 0.f;
            }
        v = s / (float)(p * p);
    }
    featH[(size_t)li * SP * SP + pos] = (_Float16)v;
}

// ---------------------------------------------------------------------------
// Stage 2: pack F^T into f16 A-matrices, zero-padded to 48(M=j) x 64(K=i).
// ---------------------------------------------------------------------------
__global__ void prep_filters(const float* __restrict__ filters, _Float16* __restrict__ filtA)
{
    int idx = blockIdx.x * blockDim.x + threadIdx.x;
    if (idx >= NKS * 48 * 64) return;
    int fi = idx / (48 * 64);
    int r  = idx % (48 * 64);
    int j = r / 64, i = r % 64;
    const int ks[NKS] = {32, 35, 38, 41, 44, 47};
    int k = ks[fi];
    float v = (j < k && i < k) ? filters[(size_t)fi * 47 * 47 + i * 47 + j] : 0.f;
    filtA[idx] = (_Float16)v;
}

// ---------------------------------------------------------------------------
// Stage 3: WMMA conv with TDM-staged B operand.
//   C[j,w'] = sum_i F^T[j,i] * feat[h*k+i, w']   (WMMA GEMM, f32 accum)
//   out[h,w] = sum_j C[j, w+j]                    (diagonal reduce in LDS)
// ---------------------------------------------------------------------------
__device__ __forceinline__ int kmap(int e, int hsel)
{
    int v = e >> 1, pos = e & 1;
    return (v < 4) ? (hsel * 8 + v * 2 + pos) : (16 + hsel * 8 + (v - 4) * 2 + pos);
}

__global__ __launch_bounds__(128) void conv_wmma(
    const _Float16* __restrict__ featH, const _Float16* __restrict__ filtA,
    float* __restrict__ outbuf, PairTab pt)
{
    __shared__ _Float16 ldsF[64][CW];   // staged B tile (rows >= k zeroed)
    __shared__ float    Clds[48][CW];   // GEMM result C[j][w']
    int p = blockIdx.z;
    int h = blockIdx.y;
    int k = pt.k[p], HF = pt.HF[p], Hout = pt.Hout[p], Wout = pt.Wout[p];
    if (h >= Hout) return;
    int w0 = blockIdx.x * NCHUNK;
    if (w0 >= Wout) return;

    int li = p % 3, fi = p / 3;
    const _Float16* feat = featH + (size_t)li * SP * SP;
    const _Float16* A    = filtA + (size_t)fi * 48 * 64;

    int lane = threadIdx.x & 31;
    int wave = threadIdx.x >> 5;
    int hsel = lane >> 4;
    int mn   = lane & 15;
    int rb   = h * k;

#if __has_builtin(__builtin_amdgcn_tensor_load_to_lds)
    // --- TDM: one 2D descriptor copies feat[rb..rb+k) x [w0, w0+CW) -> LDS.
    // OOB columns (>= HF) are hardware zero-filled via tensor_dim0.
    if (wave == 0) {
        unsigned long long gaddr =
            (unsigned long long)(size_t)(feat + (size_t)rb * SP + (size_t)w0);
        unsigned ldsOff = (unsigned)(size_t)(void*)&ldsF[0][0]; // LDS aperture: low 32 = offset
        unsigned td0 = (unsigned)(HF - w0);   // remaining row width -> zero-fill past it
        unsigned td1 = (unsigned)k;
        u32x4 g0;
        g0[0] = 1u;                                            // count=1, user mode
        g0[1] = ldsOff;                                        // lds_addr
        g0[2] = (unsigned)(gaddr & 0xffffffffu);               // global_addr[31:0]
        g0[3] = (unsigned)((gaddr >> 32) & 0x01ffffffu) | (2u << 30); // [56:32] | type=2
        i32x8 g1;
        g1[0] = (int)(1u << 16);                               // data_size=1 (2 bytes)
        g1[1] = (int)((td0 & 0xffffu) << 16);                  // tensor_dim0[15:0]
        g1[2] = (int)((td0 >> 16) | ((td1 & 0xffffu) << 16));  // dim0[31:16] | dim1[15:0]
        g1[3] = (int)((td1 >> 16) | ((unsigned)CW << 16));     // dim1[31:16] | tile_dim0
        g1[4] = (int)(td1 & 0xffffu);                          // tile_dim1 (tile_dim2=0)
        g1[5] = (int)SP;                                       // tensor_dim0_stride[31:0]
        g1[6] = 0;                                             // stride hi | dim1_stride lo
        g1[7] = 0;
        i32x4 gz = {0, 0, 0, 0};
#if defined(__clang_major__) && (__clang_major__ >= 23)
        i32x8 gz8 = {0, 0, 0, 0, 0, 0, 0, 0};
        __builtin_amdgcn_tensor_load_to_lds(g0, g1, gz, gz, gz8, 0);
#else
        __builtin_amdgcn_tensor_load_to_lds(g0, g1, gz, gz, 0);
#endif
    }
    // Zero K-padding rows [k,64) while the DMA runs (disjoint LDS region).
    for (int idx = threadIdx.x; idx < (64 - k) * CW; idx += (int)blockDim.x)
        (&ldsF[k][0])[idx] = (_Float16)0.0f;
#else
    // Fallback: stage tile with plain loads (zero-fill OOB + pad rows).
    for (int idx = threadIdx.x; idx < 64 * CW; idx += (int)blockDim.x) {
        int i = idx / CW, c = idx % CW;
        int col = w0 + c;
        _Float16 v = (_Float16)0.0f;
        if (i < k && col < HF) v = feat[(size_t)(rb + i) * SP + col];
        ldsF[i][c] = v;
    }
#endif

    // A fragments (F^T padded): 3 M-tiles x 2 K-tiles, reused across all N tiles.
    v16h Af[3][2];
    for (int mt = 0; mt < 3; ++mt)
        for (int kt = 0; kt < 2; ++kt) {
            v16h a;
#pragma unroll
            for (int e = 0; e < 16; ++e) {
                int i = kt * 32 + kmap(e, hsel);
                int j = mt * 16 + mn;
                a[e] = A[j * 64 + i];
            }
            Af[mt][kt] = a;
        }

#if __has_builtin(__builtin_amdgcn_tensor_load_to_lds)
#if __has_builtin(__builtin_amdgcn_s_wait_tensorcnt)
    if (wave == 0) __builtin_amdgcn_s_wait_tensorcnt(0);
#endif
#endif
    __syncthreads();   // publish staged tile (+ pad rows) to all waves

    for (int t = wave; t < NTILES; t += 4) {
        int c = t * 16 + mn;
        v16h Bf[2];
        for (int kt = 0; kt < 2; ++kt) {
            v16h b;
#pragma unroll
            for (int e = 0; e < 16; ++e) {
                int i = kt * 32 + kmap(e, hsel);
                b[e] = ldsF[i][c];          // unguarded: TDM zero-fill + zeroed pad rows
            }
            Bf[kt] = b;
        }
        for (int mt = 0; mt < 3; ++mt) {
            v8f acc = {};
            acc = __builtin_amdgcn_wmma_f32_16x16x32_f16(false, Af[mt][0], false, Bf[0],
                                                         (short)0, acc, false, false);
            acc = __builtin_amdgcn_wmma_f32_16x16x32_f16(false, Af[mt][1], false, Bf[1],
                                                         (short)0, acc, false, false);
#pragma unroll
            for (int r = 0; r < 8; ++r)                 // D layout: row = r + 8*half
                Clds[mt * 16 + r + 8 * hsel][t * 16 + mn] = acc[r];
        }
    }
    __syncthreads();

    int x = threadIdx.x;
    int w = w0 + x;
    if (x < NCHUNK && w < Wout) {
        float s = 0.f;
        for (int j = 0; j < k; ++j) s += Clds[j][x + j];
        outbuf[(size_t)p * OUT_SLOT + (size_t)h * Wout + w] = s;
    }
}

// ---------------------------------------------------------------------------
// Stage 4: lower median (torch-style (n-1)/2) via 32-step radix select.
// ---------------------------------------------------------------------------
__global__ __launch_bounds__(256) void median_select(
    const float* __restrict__ outbuf, float* __restrict__ med, PairTab pt)
{
    __shared__ unsigned int red[256];
    int p = blockIdx.x;
    int n = pt.Hout[p] * pt.Wout[p];
    unsigned int m = (unsigned int)((n - 1) / 2);
    const float* src = outbuf + (size_t)p * OUT_SLOT;
    unsigned int prefix = 0;
    for (int b = 31; b >= 0; --b) {
        unsigned int t = prefix | (1u << b);
        unsigned int local = 0;
        for (int idx = threadIdx.x; idx < n; idx += blockDim.x)
            if (__float_as_uint(src[idx]) < t) local++;
        red[threadIdx.x] = local;
        __syncthreads();
        for (int s = 128; s > 0; s >>= 1) {
            if ((int)threadIdx.x < s) red[threadIdx.x] += red[threadIdx.x + s];
            __syncthreads();
        }
        unsigned int total = red[0];
        __syncthreads();
        if (total <= m) prefix = t;   // keep largest t with count(v < t) <= m
    }
    if (threadIdx.x == 0) med[p] = __uint_as_float(prefix);
}

// ---------------------------------------------------------------------------
// Stage 5: write cov (thresholded, scaled column sums) and rep (column sums).
// ---------------------------------------------------------------------------
__global__ void finalize(const float* __restrict__ outbuf, const float* __restrict__ med,
                         float* __restrict__ out, PairTab pt)
{
    int p = blockIdx.y;
    int w = blockIdx.x * blockDim.x + threadIdx.x;
    int Wout = pt.Wout[p], Hout = pt.Hout[p];
    if (w >= Wout) return;
    const float* src = outbuf + (size_t)p * OUT_SLOT;
    float mv = med[p];
    float sAll = 0.f, sCov = 0.f;
    for (int h = 0; h < Hout; ++h) {
        float v = src[(size_t)h * Wout + w];
        sAll += v;
        if (v > mv) sCov += v;
    }
    const float inv_sqrt2 = 0.70710678118654752440f;
    out[pt.covoff[p] + w] = sCov * inv_sqrt2 / (1.1f * (float)Hout);
    out[pt.repoff[p] + w] = sAll;
}

// ---------------------------------------------------------------------------
extern "C" void kernel_launch(void* const* d_in, const int* in_sizes, int n_in,
                              void* d_out, int out_size, void* d_ws, size_t ws_size,
                              hipStream_t stream)
{
    (void)in_sizes; (void)n_in; (void)out_size; (void)ws_size;
    const float* S       = (const float*)d_in[0];
    const float* filters = (const float*)d_in[1];
    float* out = (float*)d_out;

    // workspace carve
    char* ws = (char*)d_ws;
    _Float16* featH = (_Float16*)ws;                         // 3 * SP*SP f16
    size_t off = (size_t)3 * SP * SP * sizeof(_Float16);
    off = (off + 255) & ~(size_t)255;
    _Float16* filtA = (_Float16*)(ws + off);                 // 6 * 48*64 f16
    off += (size_t)NKS * 48 * 64 * sizeof(_Float16);
    off = (off + 255) & ~(size_t)255;
    float* outbuf = (float*)(ws + off);                      // 18 * OUT_SLOT f32
    off += (size_t)NPAIR * OUT_SLOT * sizeof(float);
    float* med = (float*)(ws + off);                         // 18 f32

    const int ks[NKS] = {32, 35, 38, 41, 44, 47};
    PairTab pt;
    int covacc = 0;
    for (int fi = 0; fi < NKS; ++fi)
        for (int li = 0; li < 3; ++li) {
            int p = fi * 3 + li;
            int k = ks[fi], HF = SP - li;
            pt.k[p] = k;  pt.HF[p] = HF;
            pt.Hout[p] = (HF - k) / k + 1;
            pt.Wout[p] = HF - k + 1;
            pt.covoff[p] = covacc;  covacc += pt.Wout[p];
        }
    for (int p = 0; p < NPAIR; ++p) pt.repoff[p] = covacc + pt.covoff[p];

    dim3 g1((SP * SP + 255) / 256, 3);
    build_feats<<<g1, 256, 0, stream>>>(S, featH);
    prep_filters<<<(NKS * 48 * 64 + 255) / 256, 256, 0, stream>>>(filters, filtA);

    dim3 g2(9 /*chunks of 128 covering Wout<=1095*/, 35 /*max Hout*/, NPAIR);
    conv_wmma<<<g2, 128, 0, stream>>>(featH, filtA, outbuf, pt);

    median_select<<<NPAIR, 256, 0, stream>>>(outbuf, med, pt);

    dim3 g4((1095 + 255) / 256, NPAIR);
    finalize<<<g4, 256, 0, stream>>>(outbuf, med, out, pt);
}
